// HyperbolicGATLayer_65438121721894
// MI455X (gfx1250) — compile-verified, matrix operand order checked by reading, not run on hardware
//
#include <hip/hip_runtime.h>
#include <hip/hip_bf16.h>
#include <math.h>

// ---------------------------------------------------------------------------
// GAT layer for MI455X (gfx1250, wave32).
//  - Dense projections: v_wmma_f32_16x16x32_bf16 with register-blocked A reuse
//    (one wave holds the 16xK activation fragments, loops over all W tiles).
//  - Edge softmax/aggregation: hardware float atomics (global_atomic_add_f32 /
//    int-mapped atomic max), flat massive-parallel scatter.
// ---------------------------------------------------------------------------

#define N_NODES 50000
#define N_EDGES 800000
#define IN_F    128
#define OUT_F   64
#define HEADS   2
#define D_HID   (OUT_F * HEADS)   // 128
#define EPS_F   1e-8f

typedef __attribute__((ext_vector_type(16))) __bf16 v16bf;
typedef __attribute__((ext_vector_type(8)))  float  v8f;

// Load one 16x32 bf16 WMMA fragment from a row-major f32 matrix (ld = 128).
// Per-lane element layout for 16-bit A/B fragments (ISA 7.12.2):
//   lanes 0-15  : K = kb + {0..7}  and kb + {16..23}
//   lanes 16-31 : K = kb + {8..15} and kb + {24..31}
__device__ __forceinline__ v16bf load_frag_f32_as_bf16(const float* __restrict__ base,
                                                       int row, int kb, int lane) {
    const int klo = (lane < 16) ? 0 : 8;
    const float* p = base + (long)row * IN_F + kb + klo;
    v16bf f;
#pragma unroll
    for (int i = 0; i < 8; ++i) f[i] = (__bf16)p[i];
#pragma unroll
    for (int i = 0; i < 8; ++i) f[8 + i] = (__bf16)p[16 + i];
    return f;
}

// out[M, out_ld] = A[M,128] @ W[out_ld,128]^T + bias.
// One wave per 16-row M tile; A fragments (K=128) loaded ONCE into registers,
// then reused across all `ntiles` 16-column tiles of W (register blocking).
// block = 32 threads = exactly one wave -> EXEC all ones for WMMA.
__global__ __launch_bounds__(32) void wmma_linear_tiles(const float* __restrict__ A,
                                                        const float* __restrict__ W,
                                                        const float* __restrict__ bias,
                                                        float* __restrict__ out,
                                                        int ntiles, int out_ld) {
    const int lane = threadIdx.x & 31;
    const int m0 = blockIdx.x * 16;

    // A fragments for the whole K=128 reduction: 4 x v16bf = 32 VGPRs, held live.
    v16bf a[4];
#pragma unroll
    for (int f = 0; f < 4; ++f)
        a[f] = load_frag_f32_as_bf16(A, m0 + (lane & 15), f * 32, lane);

    const int mbase = m0 + ((lane < 16) ? 0 : 8);

    for (int t = 0; t < ntiles; ++t) {
        const int n0 = t * 16;
        v8f c = {};
#pragma unroll
        for (int f = 0; f < 4; ++f) {
            // B fragment: B[k][n] = W[n][k]  (W is [out,in] row-major)
            v16bf b = load_frag_f32_as_bf16(W, n0 + (lane & 15), f * 32, lane);
            c = __builtin_amdgcn_wmma_f32_16x16x32_bf16(
                    /*neg_a=*/false, a[f], /*neg_b=*/false, b,
                    /*c_mod=*/(short)0, c, /*reuse_a=*/false, /*reuse_b=*/false);
        }
        // C/D layout: VGPR r -> M = r (+8 for lanes 16-31); N = lane & 15.
        const int ncol = n0 + (lane & 15);
        const float bv = bias[ncol];
#pragma unroll
        for (int r = 0; r < 8; ++r)
            out[(long)(mbase + r) * out_ld + ncol] = c[r] + bv;
    }
}

// ---------------------------------------------------------------------------

__global__ void init_buffers(float* __restrict__ hagg,
                             float* __restrict__ m,
                             float* __restrict__ denom) {
    const int i = blockIdx.x * blockDim.x + threadIdx.x;
    if (i < N_NODES * D_HID) hagg[i] = 0.0f;
    if (i < N_NODES * HEADS) { m[i] = -INFINITY; denom[i] = 0.0f; }
}

// float atomic-max via monotone int mapping (works for mixed signs, -inf init)
__device__ __forceinline__ void atomicMaxFloat(float* addr, float val) {
    if (val >= 0.0f) atomicMax((int*)addr, __float_as_int(val));
    else             atomicMin((unsigned int*)addr, (unsigned int)__float_as_int(val));
}

// scores[e,h] = <q[dst,h,:], k[src,h,:]> / sqrt(64);  m[dst,h] = max(...)
__global__ void edge_scores(const float* __restrict__ q,
                            const float* __restrict__ k,
                            const int* __restrict__ ei,
                            float* __restrict__ sc,
                            float* __restrict__ m) {
    const int idx = blockIdx.x * blockDim.x + threadIdx.x;
    if (idx >= N_EDGES * HEADS) return;
    const int e = idx >> 1, h = idx & 1;
    const int src = ei[e];
    const int dst = ei[N_EDGES + e];
    const float4* qp = (const float4*)(q + (long)dst * D_HID + h * OUT_F);
    const float4* kp = (const float4*)(k + (long)src * D_HID + h * OUT_F);
    float s = 0.0f;
#pragma unroll
    for (int i = 0; i < OUT_F / 4; ++i) {
        float4 a = qp[i], b = kp[i];
        s += a.x * b.x + a.y * b.y + a.z * b.z + a.w * b.w;
    }
    s *= 0.125f;                 // 1/sqrt(OUT_F)
    sc[idx] = s;
    atomicMaxFloat(&m[dst * HEADS + h], s);
}

// nodes with no incoming edges: m stays -inf -> clamp to 0 (matches reference)
__global__ void fix_m(float* __restrict__ m) {
    const int i = blockIdx.x * blockDim.x + threadIdx.x;
    if (i >= N_NODES * HEADS) return;
    const float v = m[i];
    if (!isfinite(v)) m[i] = 0.0f;
}

// e = exp(score - m[dst]);  denom[dst] += e
__global__ void edge_exp(const int* __restrict__ ei,
                         const float* __restrict__ m,
                         float* __restrict__ sc,
                         float* __restrict__ denom) {
    const int idx = blockIdx.x * blockDim.x + threadIdx.x;
    if (idx >= N_EDGES * HEADS) return;
    const int e = idx >> 1, h = idx & 1;
    const int dst = ei[N_EDGES + e];
    const float val = expf(sc[idx] - m[dst * HEADS + h]);
    sc[idx] = val;
    atomicAdd(&denom[dst * HEADS + h], val);
}

// hagg[dst, j] += (e / (denom + eps)) * v[src, j]   (flat over E*128 lanes)
__global__ void edge_scatter(const float* __restrict__ v,
                             const int* __restrict__ ei,
                             const float* __restrict__ sc,
                             const float* __restrict__ denom,
                             float* __restrict__ hagg) {
    const int idx = blockIdx.x * blockDim.x + threadIdx.x;
    if (idx >= N_EDGES * D_HID) return;
    const int e = idx >> 7;         // / 128
    const int j = idx & 127;
    const int h = j >> 6;
    const int src = ei[e];
    const int dst = ei[N_EDGES + e];
    const float ev = sc[e * HEADS + h];
    const float dn = denom[dst * HEADS + h] + EPS_F;
    const float alpha = ev / dn;
    atomicAdd(&hagg[(long)dst * D_HID + j], alpha * v[(long)src * D_HID + j]);
}

// ---------------------------------------------------------------------------

extern "C" void kernel_launch(void* const* d_in, const int* in_sizes, int n_in,
                              void* d_out, int out_size, void* d_ws, size_t ws_size,
                              hipStream_t stream) {
    const float* x  = (const float*)d_in[0];
    const int*   ei = (const int*)  d_in[1];   // [2, E] int32
    const float* Wq = (const float*)d_in[2];
    const float* bq = (const float*)d_in[3];
    const float* Wk = (const float*)d_in[4];
    const float* bk = (const float*)d_in[5];
    const float* Wv = (const float*)d_in[6];
    const float* bv = (const float*)d_in[7];
    const float* Wo = (const float*)d_in[8];
    const float* bo = (const float*)d_in[9];
    float* out = (float*)d_out;

    // Workspace carve-up (floats): q,k,v,hagg [50000x128] + scores [E*2] + m,denom [50000x2]
    float* ws    = (float*)d_ws;
    float* q     = ws;
    float* kbuf  = q    + (long)N_NODES * D_HID;
    float* vbuf  = kbuf + (long)N_NODES * D_HID;
    float* hagg  = vbuf + (long)N_NODES * D_HID;
    float* sc    = hagg + (long)N_NODES * D_HID;
    float* m     = sc   + (long)N_EDGES * HEADS;
    float* denom = m    + (long)N_NODES * HEADS;

    // 0) init aggregate / softmax state
    {
        const int n = N_NODES * D_HID;
        init_buffers<<<(n + 255) / 256, 256, 0, stream>>>(hagg, m, denom);
    }

    // 1) q/k/v projections: one wave per 16-row tile, A fragments loaded once,
    //    reused across all 8 column tiles (32 WMMAs per wave).
    const int mblocks = N_NODES / 16;   // 3125, exact
    wmma_linear_tiles<<<mblocks, 32, 0, stream>>>(x, Wq, bq, q,    D_HID / 16, D_HID);
    wmma_linear_tiles<<<mblocks, 32, 0, stream>>>(x, Wk, bk, kbuf, D_HID / 16, D_HID);
    wmma_linear_tiles<<<mblocks, 32, 0, stream>>>(x, Wv, bv, vbuf, D_HID / 16, D_HID);

    // 2) per-edge scores + running segment max
    {
        const int n = N_EDGES * HEADS;
        edge_scores<<<(n + 255) / 256, 256, 0, stream>>>(q, kbuf, ei, sc, m);
    }

    // 3) clamp -inf maxima (nodes with no incoming edges)
    {
        const int n = N_NODES * HEADS;
        fix_m<<<(n + 255) / 256, 256, 0, stream>>>(m);
    }

    // 4) exponentiate + segment-sum denominator
    {
        const int n = N_EDGES * HEADS;
        edge_exp<<<(n + 255) / 256, 256, 0, stream>>>(ei, m, sc, denom);
    }

    // 5) weighted aggregation (102.4M lanes, hardware f32 atomic adds)
    {
        const int n = N_EDGES * D_HID;
        edge_scatter<<<(n + 255) / 256, 256, 0, stream>>>(vbuf, ei, sc, denom, hagg);
    }

    // 6) output projection: A reuse across 4 column tiles, writes d_out
    wmma_linear_tiles<<<mblocks, 32, 0, stream>>>(hagg, Wo, bo, out, OUT_F / 16, OUT_F);
}
